// DiffusionTransformer_67293547593950
// MI455X (gfx1250) — matrix operand order repaired and checked by reading, not compiled, not run on hardware
//
#include <hip/hip_runtime.h>
#include <cmath>

// ---------------------------------------------------------------------------
// DiffusionTransformer (AF3-style) for MI455X / gfx1250.
// Heavy GEMMs run on v_wmma_f32_16x16x32_bf16 (f32 accumulate). A tiles are
// staged fp32 into LDS via the CDNA5 async global->LDS path (ASYNCcnt) when
// the toolchain exposes it; fragments convert to bf16 at read time.
// ---------------------------------------------------------------------------

typedef __attribute__((ext_vector_type(16))) __bf16 v16bf;
typedef __attribute__((ext_vector_type(8)))  float  v8f;

union FragBF { v16bf v; __bf16 e[16]; };

__device__ __forceinline__ v8f wmma_bf16(const FragBF& a, const FragBF& b, v8f c) {
  return __builtin_amdgcn_wmma_f32_16x16x32_bf16(false, a.v, false, b.v,
                                                 (short)0, c, false, false);
}

// CDNA5 async global->LDS copy (tracked by ASYNCcnt). Guarded so the code
// still compiles (synchronous fallback) if the builtin is absent.
// Toolchain-probed signature: both pointers are generic `int __vector(4) *`.
#if defined(__HIP_DEVICE_COMPILE__)
  #if __has_builtin(__builtin_amdgcn_global_load_async_to_lds_b128)
    #define ASYNC_LDS 1
    typedef int async_v4i __attribute__((vector_size(4 * sizeof(int))));
  #endif
  #if __has_builtin(__builtin_amdgcn_s_wait_asynccnt)
    #define WAIT_ASYNC() __builtin_amdgcn_s_wait_asynccnt(0)
  #else
    #define WAIT_ASYNC() asm volatile("s_wait_asynccnt 0" ::: "memory")
  #endif
#else
  #define WAIT_ASYNC() ((void)0)
#endif

// problem constants
#define cB  2
#define cN  384
#define cD  768
#define cH  16
#define cDH 48
#define cDP 128
#define cDS 384
#define cM  (cB * cN)          // 768 token rows

// epilogue flags
#define F_SIG  1
#define F_SILU 2
#define F_ADDC 4
#define F_MULC 8

#define BK 32

// ---------------------------------------------------------------------------
// Generic batched GEMM: C[b] = epilogue( A[b] (MxK) * B[b] (KxN or NxK) )
// 128 threads = 4 waves; block tile 64x64; each wave owns a 32x32 subtile
// made of four 16x16 WMMA accumulators.
// batch offset = (z / Hdiv)*s?1 + (z % Hdiv)*s?2  (Hdiv=1 for plain batches)
// Requirement (met by all call sites): M % 64 == 0, A rows 16B aligned.
// ---------------------------------------------------------------------------
template<bool TRANSB>
__global__ __launch_bounds__(128)
void gemm_bf16_64x64(const float* __restrict__ A, const float* __restrict__ Bm,
                     float* __restrict__ C,
                     int M, int N, int K, int lda, int ldb, int ldc,
                     int Hdiv,
                     long long sA1, long long sA2,
                     long long sB1, long long sB2,
                     long long sC1, long long sC2,
                     const float* __restrict__ bias,
                     float alpha, int flags)
{
  __shared__ float  ldsAf[64 * BK];  // A tile staged fp32 (async-copy friendly)
  __shared__ __bf16 ldsB [64 * BK];  // B tile bf16, stored N-major: [n][k]

  const int z  = blockIdx.z;
  const float* Ab = A  + (long long)(z / Hdiv) * sA1 + (long long)(z % Hdiv) * sA2;
  const float* Bb = Bm + (long long)(z / Hdiv) * sB1 + (long long)(z % Hdiv) * sB2;
  float*       Cb = C  + (long long)(z / Hdiv) * sC1 + (long long)(z % Hdiv) * sC2;

  const int m0   = blockIdx.y * 64;
  const int n0   = blockIdx.x * 64;
  const int tid  = threadIdx.x;
  const int wave = tid >> 5;
  const int lane = tid & 31;
  const int wr   = wave >> 1;   // wave row (0..1) -> 32 rows
  const int wc   = wave & 1;    // wave col (0..1) -> 32 cols
  const int nlo  = lane & 15;
  const int hi   = lane >> 4;

  v8f acc[2][2];
#pragma unroll
  for (int i = 0; i < 2; ++i)
#pragma unroll
    for (int j = 0; j < 2; ++j)
#pragma unroll
      for (int r = 0; r < 8; ++r) acc[i][j][r] = 0.0f;

  for (int k0 = 0; k0 < K; k0 += BK) {
    // ---------------- stage A tile (64 x 32 fp32) ----------------
#if defined(ASYNC_LDS)
    if (K - k0 >= BK) {
      // full chunk: async DMA 16B per lane straight into LDS (no VGPR hop)
      for (int c = tid; c < 64 * (BK / 4); c += 128) {
        int m  = c >> 3;
        int kg = (c & 7) * 4;
        const float* gp = Ab + (long long)(m0 + m) * lda + (k0 + kg);
        __builtin_amdgcn_global_load_async_to_lds_b128(
            (async_v4i*)gp, (async_v4i*)&ldsAf[m * BK + kg], 0, 0);
      }
    } else
#endif
    {
      for (int i = tid; i < 64 * BK; i += 128) {
        int m = i >> 5, k = i & 31;
        int gm = m0 + m, gk = k0 + k;
        ldsAf[m * BK + k] = (gm < M && gk < K) ? Ab[(long long)gm * lda + gk] : 0.0f;
      }
    }

    // ---------------- stage B tile, bf16, N-major [n][k] ----------------
    if (TRANSB) {                     // B given as N x K (row-major)
      for (int i = tid; i < 64 * BK; i += 128) {
        int n = i >> 5, k = i & 31;
        int gn = n0 + n, gk = k0 + k;
        float v = (gn < N && gk < K) ? Bb[(long long)gn * ldb + gk] : 0.0f;
        ldsB[n * BK + k] = (__bf16)v;
      }
    } else {                          // B given as K x N (row-major)
      for (int i = tid; i < 64 * BK; i += 128) {
        int k = i >> 6, n = i & 63;
        int gn = n0 + n, gk = k0 + k;
        float v = (gn < N && gk < K) ? Bb[(long long)gk * ldb + gn] : 0.0f;
        ldsB[n * BK + k] = (__bf16)v;
      }
    }

    // prefetch next K chunk while this one is consumed
    if (k0 + BK < K) {
      __builtin_prefetch(Ab + (long long)(m0 + (tid >> 1)) * lda + (k0 + BK), 0, 1);
      __builtin_prefetch(Bb + (long long)(tid) * ldb + (TRANSB ? (k0 + BK) : n0), 0, 1);
    }

#if defined(ASYNC_LDS)
    WAIT_ASYNC();                     // wave's async LDS writes complete
#endif
    __syncthreads();                  // cross-wave visibility

    // --- load fragments (ISA 7.12.2 layouts) and issue 4 WMMAs ---
    FragBF fa[2], fb[2];
#pragma unroll
    for (int s = 0; s < 2; ++s) {
      const float* base = &ldsAf[(wr * 32 + s * 16 + nlo) * BK];
      // A 16x32 bf16: lanes 0-15 rows M=lane K=0..7,16..23; lanes 16-31 K=8..15,24..31
#pragma unroll
      for (int t = 0; t < 8; ++t) fa[s].e[t]     = (__bf16)base[hi * 8 + t];
#pragma unroll
      for (int t = 0; t < 8; ++t) fa[s].e[8 + t] = (__bf16)base[16 + hi * 8 + t];
    }
#pragma unroll
    for (int s = 0; s < 2; ++s) {
      const __bf16* base = &ldsB[(wc * 32 + s * 16 + nlo) * BK];
      // B 32x16 bf16: lanes 0-15 K=0..7,16..23; lanes 16-31 K=8..15,24..31 (N = lane&15)
#pragma unroll
      for (int t = 0; t < 8; ++t) fb[s].e[t]     = base[hi * 8 + t];
#pragma unroll
      for (int t = 0; t < 8; ++t) fb[s].e[8 + t] = base[16 + hi * 8 + t];
    }
#pragma unroll
    for (int i = 0; i < 2; ++i)
#pragma unroll
      for (int j = 0; j < 2; ++j)
        acc[i][j] = wmma_bf16(fa[i], fb[j], acc[i][j]);
    __syncthreads();
  }

  // --- epilogue: C/D layout, lane: n = lane&15, vgpr r: m = r + 8*(lane>>4) ---
#pragma unroll
  for (int i = 0; i < 2; ++i) {
#pragma unroll
    for (int j = 0; j < 2; ++j) {
      int gn = n0 + wc * 32 + j * 16 + nlo;
      if (gn >= N) continue;
      float bv = bias ? bias[gn] : 0.0f;
#pragma unroll
      for (int r = 0; r < 8; ++r) {
        int gm = m0 + wr * 32 + i * 16 + r + 8 * hi;
        if (gm >= M) continue;
        float v = (acc[i][j][r] + bv) * alpha;
        if (flags & F_SIG)  v = 1.0f / (1.0f + __expf(-v));
        if (flags & F_SILU) v = v / (1.0f + __expf(-v));
        long long idx = (long long)gm * ldc + gn;
        if (flags & F_ADDC) v += Cb[idx];
        if (flags & F_MULC) v *= Cb[idx];
        Cb[idx] = v;
      }
    }
  }
}

// ---------------------------------------------------------------------------
// Pair bias: biasT[b,h,i,j] = sum_dp z[b,i,j,dp]*wb[dp,h] + bb[h] + beta[b,i,j]
// Written directly in (B,H,N,N) layout so QK^T can accumulate in place.
// One 16x16 WMMA tile per wave (rows = 16 consecutive j, cols = 16 heads).
// ---------------------------------------------------------------------------
__global__ __launch_bounds__(128)
void bias_wmma_kernel(const float* __restrict__ z, const float* __restrict__ wb,
                      const float* __restrict__ bb, const float* __restrict__ beta,
                      float* __restrict__ biasT)
{
  const int wave = threadIdx.x >> 5;
  const int lane = threadIdx.x & 31;
  const int nlo  = lane & 15;
  const int hi   = lane >> 4;
  const long long r0 = ((long long)blockIdx.x * 4 + wave) * 16;  // row = (b,i,j)

  v8f acc;
#pragma unroll
  for (int r = 0; r < 8; ++r) acc[r] = 0.0f;

  const float* zrow = z + (r0 + nlo) * cDP;   // A: M = lane&15 for both halves
  for (int k0 = 0; k0 < cDP; k0 += BK) {
    FragBF fa, fb;
#pragma unroll
    for (int t = 0; t < 8; ++t) fa.e[t]     = (__bf16)zrow[k0 + hi * 8 + t];
#pragma unroll
    for (int t = 0; t < 8; ++t) fa.e[8 + t] = (__bf16)zrow[k0 + 16 + hi * 8 + t];
#pragma unroll
    for (int t = 0; t < 8; ++t) fb.e[t]     = (__bf16)wb[(k0 + hi * 8 + t) * cH + nlo];
#pragma unroll
    for (int t = 0; t < 8; ++t) fb.e[8 + t] = (__bf16)wb[(k0 + 16 + hi * 8 + t) * cH + nlo];
    acc = wmma_bf16(fa, fb, acc);
  }

  const int j0 = (int)(r0 % cN);
  const int ii = (int)((r0 / cN) % cN);
  const int b  = (int)(r0 / ((long long)cN * cN));
  const float bbv = bb[nlo];
  float*       outb = biasT + (((long long)(b * cH + nlo) * cN + ii) * cN + j0);
  const float* betr = beta  + ((long long)(b * cN + ii) * cN + j0);
#pragma unroll
  for (int r = 0; r < 8; ++r) {
    int m = r + 8 * hi;
    outb[m] = acc[r] + bbv + betr[m];
  }
}

// ---------------------------------------------------------------------------
// Row-wise layernorm (optional affine), one block per row.
// ---------------------------------------------------------------------------
__global__ __launch_bounds__(128)
void layernorm_kernel(const float* __restrict__ x, float* __restrict__ y,
                      int cols, const float* __restrict__ g, const float* __restrict__ b)
{
  const long long row = blockIdx.x;
  const float* xr = x + row * cols;
  float*       yr = y + row * cols;
  __shared__ float red[128], red2[128];
  const int tid = threadIdx.x;
  float s = 0.0f, s2 = 0.0f;
  for (int c = tid; c < cols; c += 128) { float v = xr[c]; s += v; s2 += v * v; }
  red[tid] = s; red2[tid] = s2;
  __syncthreads();
  for (int st = 64; st > 0; st >>= 1) {
    if (tid < st) { red[tid] += red[tid + st]; red2[tid] += red2[tid + st]; }
    __syncthreads();
  }
  const float mean = red[0] / cols;
  const float var  = red2[0] / cols - mean * mean;
  const float inv  = rsqrtf(var + 1e-5f);
  for (int c = tid; c < cols; c += 128) {
    float v = (xr[c] - mean) * inv;
    if (g) v = v * g[c] + b[c];
    yr[c] = v;
  }
}

// ---------------------------------------------------------------------------
// In-place softmax over the last dim (key dim j), one block per row.
// ---------------------------------------------------------------------------
__global__ __launch_bounds__(128)
void softmax_kernel(float* __restrict__ S, int cols)
{
  float* p = S + (long long)blockIdx.x * cols;
  __shared__ float red[128];
  const int tid = threadIdx.x;
  float m = -1e30f;
  for (int c = tid; c < cols; c += 128) m = fmaxf(m, p[c]);
  red[tid] = m; __syncthreads();
  for (int st = 64; st > 0; st >>= 1) {
    if (tid < st) red[tid] = fmaxf(red[tid], red[tid + st]);
    __syncthreads();
  }
  m = red[0]; __syncthreads();
  float sum = 0.0f;
  for (int c = tid; c < cols; c += 128) { float e = __expf(p[c] - m); p[c] = e; sum += e; }
  red[tid] = sum; __syncthreads();
  for (int st = 64; st > 0; st >>= 1) {
    if (tid < st) red[tid] += red[tid + st];
    __syncthreads();
  }
  const float inv = 1.0f / red[0];
  for (int c = tid; c < cols; c += 128) p[c] *= inv;
}

// ---------------------------------------------------------------------------
// Elementwise glue.
// ---------------------------------------------------------------------------
__global__ void ew_mul_kernel(float* __restrict__ y, const float* __restrict__ x, int n) {
  int i = blockIdx.x * 256 + threadIdx.x; if (i < n) y[i] *= x[i];
}
__global__ void ew_fma_kernel(float* __restrict__ o, const float* __restrict__ sg,
                              const float* __restrict__ an, const float* __restrict__ ln, int n) {
  int i = blockIdx.x * 256 + threadIdx.x; if (i < n) o[i] = sg[i] * an[i] + ln[i];
}
__global__ void ew_add_kernel(float* __restrict__ o, const float* __restrict__ x,
                              const float* __restrict__ y, int n) {
  int i = blockIdx.x * 256 + threadIdx.x; if (i < n) o[i] = x[i] + y[i];
}

// ---------------------------------------------------------------------------
// Host side
// ---------------------------------------------------------------------------
struct BP {
  const float *ln_g, *ln_b, *w_s, *w_s_b, *b_s;                       // att adaln
  const float *wq, *bq, *wk, *wv, *wg, *wo, *wb, *bb, *ws, *bs;      // att
  const float *t_ln_g, *t_ln_b, *t_w_s, *t_w_s_b, *t_b_s;            // trans adaln
  const float *w1, *w2, *w3, *twg, *tbg;                             // trans
};

static void launch_gemm(bool transB,
                        const float* A, const float* Bm, float* C,
                        int M, int N, int K, int lda, int ldb, int ldc,
                        int batch, int Hdiv,
                        long long sA1, long long sA2,
                        long long sB1, long long sB2,
                        long long sC1, long long sC2,
                        const float* bias, float alpha, int flags, hipStream_t stream)
{
  dim3 grid((N + 63) / 64, (M + 63) / 64, batch);
  if (transB)
    gemm_bf16_64x64<true><<<grid, 128, 0, stream>>>(A, Bm, C, M, N, K, lda, ldb, ldc,
        Hdiv, sA1, sA2, sB1, sB2, sC1, sC2, bias, alpha, flags);
  else
    gemm_bf16_64x64<false><<<grid, 128, 0, stream>>>(A, Bm, C, M, N, K, lda, ldb, ldc,
        Hdiv, sA1, sA2, sB1, sB2, sC1, sC2, bias, alpha, flags);
}

extern "C" void kernel_launch(void* const* d_in, const int* in_sizes, int n_in,
                              void* d_out, int out_size, void* d_ws, size_t ws_size,
                              hipStream_t stream)
{
  const float* a_in    = (const float*)d_in[0];
  const float* s_in    = (const float*)d_in[1];
  const float* z_in    = (const float*)d_in[2];
  const float* beta_in = (const float*)d_in[3];

  // Detect param flatten order: jax tree_flatten (sorted keys) vs dict insertion.
  const bool sorted = (n_in > 4) && (in_sizes[4] == cDS * cD);  // first leaf = att.adaln.b_s

  // workspace carve-out (fp32)
  float* w = (float*)d_ws;
  auto alloc = [&](size_t n) { float* p = w; w += n; return p; };
  const size_t MD = (size_t)cM * cD;          // 589824
  float* a_ping = alloc(MD);
  float* a_pong = alloc(MD);
  float* an     = alloc(MD);
  float* sn     = alloc((size_t)cM * cDS);
  float* sig    = alloc(MD);
  float* lin    = alloc(MD);
  float* a2     = alloc(MD);
  float* qb     = alloc(MD);
  float* kb     = alloc(MD);
  float* vb     = alloc(MD);
  float* gb     = alloc(MD);
  float* scores = alloc((size_t)cB * cH * cN * cN);   // 4718592
  float* ob     = alloc(MD);
  float* attn   = alloc(MD);
  float* buf1   = alloc((size_t)cM * 2 * cD);
  float* buf2   = alloc((size_t)cM * 2 * cD);
  float* tout   = alloc(MD);
  (void)ws_size;

  const float qscale = 1.0f / sqrtf((float)cDH);
  const long long sQA = (long long)cN * cD;       // per-b stride in (B,N,H*DH)
  const long long sSC = (long long)cH * cN * cN;  // per-b stride in (B,H,N,N)

  const float* aIn = a_in;
  for (int l = 0; l < 4; ++l) {
    const int base = 4 + 25 * l;
    auto P = [&](int i) { return (const float*)d_in[base + i]; };
    BP p;
    if (sorted) {
      p.b_s = P(0);  p.ln_b = P(1);  p.ln_g = P(2);  p.w_s = P(3);  p.w_s_b = P(4);
      p.bb = P(5);   p.bq = P(6);    p.bs = P(7);    p.wb = P(8);   p.wg = P(9);
      p.wk = P(10);  p.wo = P(11);   p.wq = P(12);   p.ws = P(13);  p.wv = P(14);
      p.t_b_s = P(15); p.t_ln_b = P(16); p.t_ln_g = P(17); p.t_w_s = P(18); p.t_w_s_b = P(19);
      p.tbg = P(20); p.w1 = P(21);   p.w2 = P(22);   p.w3 = P(23);  p.twg = P(24);
    } else {
      p.ln_g = P(0); p.ln_b = P(1);  p.w_s = P(2);   p.w_s_b = P(3); p.b_s = P(4);
      p.wq = P(5);   p.bq = P(6);    p.wk = P(7);    p.wv = P(8);    p.wg = P(9);
      p.wo = P(10);  p.wb = P(11);   p.bb = P(12);   p.ws = P(13);   p.bs = P(14);
      p.t_ln_g = P(15); p.t_ln_b = P(16); p.t_w_s = P(17); p.t_w_s_b = P(18); p.t_b_s = P(19);
      p.w1 = P(20);  p.w2 = P(21);   p.w3 = P(22);   p.twg = P(23);  p.tbg = P(24);
    }
    float* aOut = (l == 3) ? (float*)d_out : ((l & 1) ? a_pong : a_ping);

    // ---------------- attention branch ----------------
    layernorm_kernel<<<cM, 128, 0, stream>>>(aIn, an, cD, nullptr, nullptr);
    layernorm_kernel<<<cM, 128, 0, stream>>>(s_in, sn, cDS, p.ln_g, p.ln_b);
    launch_gemm(false, sn, p.w_s, sig, cM, cD, cDS, cDS, cD, cD, 1, 1, 0,0,0,0,0,0,
                p.w_s_b, 1.0f, F_SIG, stream);
    launch_gemm(false, sn, p.b_s, lin, cM, cD, cDS, cDS, cD, cD, 1, 1, 0,0,0,0,0,0,
                nullptr, 1.0f, 0, stream);
    ew_fma_kernel<<<(int)((MD + 255) / 256), 256, 0, stream>>>(a2, sig, an, lin, (int)MD);

    launch_gemm(false, a2, p.wq, qb, cM, cD, cD, cD, cD, cD, 1, 1, 0,0,0,0,0,0,
                p.bq, qscale, 0, stream);
    launch_gemm(false, a2, p.wk, kb, cM, cD, cD, cD, cD, cD, 1, 1, 0,0,0,0,0,0,
                nullptr, 1.0f, 0, stream);
    launch_gemm(false, a2, p.wv, vb, cM, cD, cD, cD, cD, cD, 1, 1, 0,0,0,0,0,0,
                nullptr, 1.0f, 0, stream);
    launch_gemm(false, a2, p.wg, gb, cM, cD, cD, cD, cD, cD, 1, 1, 0,0,0,0,0,0,
                nullptr, 1.0f, F_SIG, stream);   // gb = sigmoid(g)

    // pair bias -> scores (B,H,N,N)
    bias_wmma_kernel<<<(cB * cN * cN) / 64, 128, 0, stream>>>(z_in, p.wb, p.bb, beta_in, scores);

    // scores += Q K^T   (batched over b,h; B side is "N x K" = k rows)
    launch_gemm(true, qb, kb, scores, cN, cN, cDH, cD, cD, cN,
                cB * cH, cH, sQA, cDH, sQA, cDH, sSC, (long long)cN * cN,
                nullptr, 1.0f, F_ADDC, stream);
    softmax_kernel<<<cB * cH * cN, 128, 0, stream>>>(scores, cN);

    // O = P V  (batched; output scattered into (B,N,H,DH))
    launch_gemm(false, scores, vb, ob, cN, cDH, cN, cN, cD, cD,
                cB * cH, cH, sSC, (long long)cN * cN, sQA, cDH, sQA, cDH,
                nullptr, 1.0f, 0, stream);

    ew_mul_kernel<<<(int)((MD + 255) / 256), 256, 0, stream>>>(ob, gb, (int)MD);

    // attn = sigmoid(s@ws + bs); then attn = (o @ wo) * attn
    launch_gemm(false, s_in, p.ws, attn, cM, cD, cDS, cDS, cD, cD, 1, 1, 0,0,0,0,0,0,
                p.bs, 1.0f, F_SIG, stream);
    launch_gemm(false, ob, p.wo, attn, cM, cD, cD, cD, cD, cD, 1, 1, 0,0,0,0,0,0,
                nullptr, 1.0f, F_MULC, stream);

    // ---------------- transition branch ----------------
    layernorm_kernel<<<cM, 128, 0, stream>>>(s_in, sn, cDS, p.t_ln_g, p.t_ln_b);
    launch_gemm(false, sn, p.t_w_s, sig, cM, cD, cDS, cDS, cD, cD, 1, 1, 0,0,0,0,0,0,
                p.t_w_s_b, 1.0f, F_SIG, stream);
    launch_gemm(false, sn, p.t_b_s, lin, cM, cD, cDS, cDS, cD, cD, 1, 1, 0,0,0,0,0,0,
                nullptr, 1.0f, 0, stream);
    ew_fma_kernel<<<(int)((MD + 255) / 256), 256, 0, stream>>>(a2, sig, an, lin, (int)MD);

    launch_gemm(false, a2, p.w1, buf1, cM, 2 * cD, cD, cD, 2 * cD, 2 * cD, 1, 1, 0,0,0,0,0,0,
                nullptr, 1.0f, F_SILU, stream);
    launch_gemm(false, a2, p.w2, buf2, cM, 2 * cD, cD, cD, 2 * cD, 2 * cD, 1, 1, 0,0,0,0,0,0,
                nullptr, 1.0f, 0, stream);
    ew_mul_kernel<<<(int)((2 * MD + 255) / 256), 256, 0, stream>>>(buf1, buf2, (int)(2 * MD));

    // tout = sigmoid(s@wg + bg); tout = (buf1 @ w3) * tout
    launch_gemm(false, s_in, p.twg, tout, cM, cD, cDS, cDS, cD, cD, 1, 1, 0,0,0,0,0,0,
                p.tbg, 1.0f, F_SIG, stream);
    launch_gemm(false, buf1, p.w3, tout, cM, cD, 2 * cD, 2 * cD, cD, cD, 1, 1, 0,0,0,0,0,0,
                nullptr, 1.0f, F_MULC, stream);

    // a = attn + transition
    ew_add_kernel<<<(int)((MD + 255) / 256), 256, 0, stream>>>(aOut, attn, tout, (int)MD);
    aIn = aOut;
  }
  (void)out_size; (void)n_in;
}
